// FullGraphModel_11416023073436
// MI455X (gfx1250) — matrix operand order repaired and checked by reading, not compiled, not run on hardware
//
#include <hip/hip_runtime.h>
#include <hip/hip_bf16.h>

typedef __attribute__((ext_vector_type(16))) _Float16 v16h;
typedef __attribute__((ext_vector_type(8)))  float    v8f;

#define NN 131072
#define TT 128
#define CC 1024
#define EE (NN*32)
#define EPSLN 1e-5f

#define LOG2E    1.4426950408889634f
#define LN2      0.6931471805599453f

// ---------- helpers ----------
// Element index of retina_w fits in 32 bits (2^27 elements total).
__device__ __forceinline__ float hash_u01(unsigned x) {
  x *= 0x9E3779B9u; x ^= x >> 16; x *= 0x7FEB352Du; x ^= x >> 15; x *= 0x846CA68Bu; x ^= x >> 16;
  return (float)(x >> 8) * (1.0f/16777216.0f) + 1e-7f;
}

// exp(w + gumbel(idx)) with gumbel = -ln(-ln u)  =>  exp(w) / (-ln u)
// = exp2(w*log2e) * rcp(-log2(u)*ln2)   : exactly 3 TRANS ops (exp2, log2, rcp)
__device__ __forceinline__ float sm_num(float w, unsigned idx) {
  float u  = hash_u01(idx);
  float nl = -__builtin_amdgcn_logf(u) * LN2;           // -ln u  (v_log_f32)
  return __builtin_amdgcn_exp2f(w * LOG2E)              // exp(w) (v_exp_f32)
       * __builtin_amdgcn_rcpf(nl);                     // / z    (v_rcp_f32)
}

__device__ __forceinline__ float wredf(float v) {
  #pragma unroll
  for (int o = 16; o > 0; o >>= 1) v += __shfl_xor(v, o, 32);
  return v;
}

__device__ __forceinline__ unsigned f2ord(float f) {
  unsigned u = __float_as_uint(f);
  return (u & 0x80000000u) ? ~u : (u | 0x80000000u);
}
__device__ __forceinline__ float ord2f(unsigned u) {
  return (u & 0x80000000u) ? __uint_as_float(u ^ 0x80000000u) : __uint_as_float(~u);
}

// ---------- scalar workspace init ----------
__global__ void k_init(float* sc) {
  int i = threadIdx.x;
  if (i < 64) sc[i] = 0.0f;
  __syncthreads();
  if (i == 0) { ((unsigned*)sc)[34] = 0xFFFFFFFFu; ((unsigned*)sc)[35] = 0u; }
}

// ---------- pass 1: per-(t,j) softmax denominators (column sums) ----------
__global__ void k_colsum(const float* __restrict__ W, float* __restrict__ rden) {
  int j = blockIdx.x * blockDim.x + threadIdx.x;     // column
  int t = blockIdx.y;
  const float* col = W + (size_t)t * CC * CC + j;
  unsigned base = (unsigned)t * CC * CC + (unsigned)j;
  float s = 0.0f;
  #pragma unroll 4
  for (int i = 0; i < CC; ++i) {
    __builtin_prefetch(col + (size_t)(i + 24) * CC, 0, 1);   // global_prefetch_b8
    float w = col[(size_t)i * CC];
    s += sm_num(w, base + (unsigned)i * CC);
  }
  rden[t * CC + j] = __builtin_amdgcn_rcpf(s);
}

// ---------- pass 2: x[t,i] = sum_j softmax(w)[t,i,j] * xr[t,j]  via WMMA ----------
// wave handles a 16-row tile; A = fp16 probs in ISA lane layout, B broadcasts xr
// across all 16 N-columns; D column 0 = matvec result.
__global__ void __launch_bounds__(256)
k_matvec_wmma(const float* __restrict__ W, const float* __restrict__ x0,
              const float* __restrict__ rden, float* __restrict__ xout) {
  __shared__ float s_x[CC];
  __shared__ float s_rd[CC];
  int t = blockIdx.x;
  int rowBase = blockIdx.y * 128;                 // 8 waves * 16 rows
  for (int j = threadIdx.x; j < CC; j += 256) {
    s_x[j]  = x0[t * CC + j];
    s_rd[j] = rden[t * CC + j];
  }
  __syncthreads();
  int wave = threadIdx.x >> 5;
  int lane = threadIdx.x & 31;
  int hi   = lane >> 4;                           // half-wave select
  int i0   = rowBase + wave * 16;
  int row  = i0 + (lane & 15);
  const float* Wrow = W + (size_t)t * CC * CC + (size_t)row * CC;
  unsigned bidx = ((unsigned)t * CC + (unsigned)row) * CC;
  v8f acc = {};
  #pragma unroll 2
  for (int j0 = 0; j0 < CC; j0 += 32) {
    v16h a, b;
    #pragma unroll
    for (int m = 0; m < 16; ++m) {
      int q = m >> 1, h2 = m & 1;
      // 16-bit A layout: lane<16 -> K in {0..7, 16..23}; lane>=16 -> +8
      int K = ((q < 4) ? (2*q + h2) : (16 + 2*(q - 4) + h2)) + hi * 8;
      float w = Wrow[j0 + K];
      float p = sm_num(w, bidx + (unsigned)(j0 + K)) * s_rd[j0 + K];
      a[m] = (_Float16)p;
      // 16-bit B layout: same as A: lane<16 -> K in {0..7, 16..23}; lane>=16 -> +8
      b[m] = (_Float16)s_x[j0 + m + ((m >= 8) ? 8 : 0) + hi * 8];
    }
    acc = __builtin_amdgcn_wmma_f32_16x16x32_f16(false, a, false, b,
                                                 (short)0, acc, false, false);
  }
  // D layout: lane L, VGPR r -> M = r + 8*(L>=16), N = L&15 ; take N==0
  if ((lane & 15) == 0) {
    int mBase = i0 + hi * 8;
    #pragma unroll
    for (int r = 0; r < 8; ++r) xout[t * CC + mBase + r] = acc[r];
  }
}

// ---------- masked standardization of x ----------
__global__ void k_mask_reduce(const float* __restrict__ x, float* __restrict__ sc) {
  int i = blockIdx.x * blockDim.x + threadIdx.x;
  float v = x[i];
  float m = (v != 0.0f) ? 1.0f : 0.0f;
  float n  = wredf(m);
  float s  = wredf(m * v);
  float s2 = wredf(m * v * v);
  if ((threadIdx.x & 31) == 0) {
    atomicAdd(&sc[0], n); atomicAdd(&sc[1], s); atomicAdd(&sc[2], s2);
  }
}

__global__ void k_mask_apply(const float* __restrict__ x, const float* __restrict__ sc,
                             float* __restrict__ h) {
  int i = blockIdx.x * blockDim.x + threadIdx.x;
  float n = sc[0], s = sc[1], s2 = sc[2];
  float mean = s * __builtin_amdgcn_rcpf(n);
  float var  = (s2 - n * mean * mean) * __builtin_amdgcn_rcpf(n - 1.0f);
  float rs   = __builtin_amdgcn_rsqf(var);
  float v = x[i];
  h[i] = (v != 0.0f) ? (v - mean) * rs : v;
}

// ---------- message passing ----------
__global__ void k_scatter(const int* __restrict__ ei, const float* __restrict__ ewraw,
                          const float* __restrict__ h, float* __restrict__ agg) {
  int e = blockIdx.x * blockDim.x + threadIdx.x;
  if (e >= EE) return;
  int s = ei[e];
  int d = ei[EE + e];
  // log1p(ew) = log2(1+ew) * ln2, single v_log_f32 (ew in [0,1): well conditioned)
  float lw = __builtin_amdgcn_logf(1.0f + ewraw[e]) * LN2;
  atomicAdd(&agg[d], h[s] * lw);
}

__global__ void k_sum_reduce(const float* __restrict__ a, float* __restrict__ sc) {
  int i = blockIdx.x * blockDim.x + threadIdx.x;
  float v = a[i];
  float s  = wredf(v);
  float s2 = wredf(v * v);
  if ((threadIdx.x & 31) == 0) { atomicAdd(&sc[0], s); atomicAdd(&sc[1], s2); }
}

__global__ void k_layernorm(const float* __restrict__ agg, const float* __restrict__ sc,
                            const float* __restrict__ g, const float* __restrict__ b,
                            float* __restrict__ h) {
  int i = blockIdx.x * blockDim.x + threadIdx.x;
  float mu  = sc[0] * (1.0f / NN);
  float var = sc[1] * (1.0f / NN) - mu * mu;
  float rs  = __builtin_amdgcn_rsqf(var + EPSLN);
  h[i] = (agg[i] - mu) * rs * g[i] + b[i];
}

// ---------- final selected min/max/mean ----------
__global__ void k_sel_reduce(const float* __restrict__ h, const int* __restrict__ dmv,
                             float* __restrict__ scf, unsigned* __restrict__ scu) {
  int i = blockIdx.x * blockDim.x + threadIdx.x;
  float v = h[i];
  bool sel = (dmv[i] == 1);
  float k = wredf(sel ? 1.0f : 0.0f);
  float s = wredf(sel ? v : 0.0f);
  unsigned mn = sel ? f2ord(v) : 0xFFFFFFFFu;
  unsigned mx = sel ? f2ord(v) : 0u;
  #pragma unroll
  for (int o = 16; o > 0; o >>= 1) {
    unsigned on = (unsigned)__shfl_xor((int)mn, o, 32);
    unsigned ox = (unsigned)__shfl_xor((int)mx, o, 32);
    mn = (on < mn) ? on : mn;
    mx = (ox > mx) ? ox : mx;
  }
  if ((threadIdx.x & 31) == 0) {
    atomicAdd(&scf[0], k); atomicAdd(&scf[1], s);
    atomicMin(&scu[0], mn); atomicMax(&scu[1], mx);
  }
}

__global__ void k_final(const float* __restrict__ scf, const unsigned* __restrict__ scu,
                        const float* __restrict__ fw, const float* __restrict__ fb,
                        float* __restrict__ out) {
  float k = scf[0], s = scf[1];
  float mn = ord2f(scu[0]), mx = ord2f(scu[1]);
  float m = (s - k * mn) / ((mx - mn) * k);
  float o = m * fw[0] + fb[0];
  out[0] = o > 0.0f ? o : 0.0f;
}

// ---------- launcher ----------
extern "C" void kernel_launch(void* const* d_in, const int* in_sizes, int n_in,
                              void* d_out, int out_size, void* d_ws, size_t ws_size,
                              hipStream_t stream) {
  (void)in_sizes; (void)n_in; (void)out_size; (void)ws_size;
  const float* x_in = (const float*)d_in[0];
  const int*   ei   = (const int*)d_in[1];
  const float* ewr  = (const float*)d_in[2];
  const float* W    = (const float*)d_in[3];
  const float* lng  = (const float*)d_in[4];
  const float* lnb  = (const float*)d_in[5];
  const float* fcw  = (const float*)d_in[6];
  const float* fcb  = (const float*)d_in[7];
  const int*   dmv  = (const int*)d_in[8];

  float* rden = (float*)d_ws;         // T*C
  float* xbuf = rden + TT * CC;       // N
  float* hbuf = xbuf + NN;            // N
  float* agg  = hbuf + NN;            // N
  float* sc   = agg  + NN;            // 64 scalar slots

  k_init<<<1, 64, 0, stream>>>(sc);

  dim3 gcol(CC / 256, TT);
  k_colsum<<<gcol, 256, 0, stream>>>(W, rden);

  dim3 gmv(TT, CC / 128);
  k_matvec_wmma<<<gmv, 256, 0, stream>>>(W, x_in, rden, xbuf);

  k_mask_reduce<<<NN / 256, 256, 0, stream>>>(xbuf, sc);
  k_mask_apply <<<NN / 256, 256, 0, stream>>>(xbuf, sc, hbuf);

  for (int p = 0; p < 4; ++p) {
    hipMemsetAsync(agg, 0, NN * sizeof(float), stream);
    k_scatter   <<<EE / 256, 256, 0, stream>>>(ei, ewr, hbuf, agg);
    k_sum_reduce<<<NN / 256, 256, 0, stream>>>(agg, sc + 8 + 2 * p);
    k_layernorm <<<NN / 256, 256, 0, stream>>>(agg, sc + 8 + 2 * p, lng, lnb, hbuf);
  }

  k_sel_reduce<<<NN / 256, 256, 0, stream>>>(hbuf, dmv, sc + 32, (unsigned*)(sc + 34));
  k_final<<<1, 1, 0, stream>>>(sc + 32, (unsigned*)(sc + 34), fcw, fcb, (float*)d_out);
}
